// mambablock_60954175865187
// MI455X (gfx1250) — compile-verified, matrix operand order checked by reading, not run on hardware
//
#include <hip/hip_runtime.h>
#include <hip/hip_bf16.h>

// ---------------------------------------------------------------------------
// Problem constants (from reference)
// ---------------------------------------------------------------------------
#define BATCHN 2
#define G0C    48
#define HH     96
#define WWC    96
#define HWSZ   (HH*WWC)          // 9216
#define LSEQ   HWSZ              // sequence length
#define DIN    96                // D_INNER
#define DSTATE 16
#define DTRANK 3
#define DBLW   35                // DT_RANK + 2*D_STATE
#define MROWS  (BATCHN*LSEQ)     // 18432  (multiple of 64; 9216 % 64 == 0)

typedef _Float16 v16h __attribute__((ext_vector_type(16)));
typedef _Float16 v8h  __attribute__((ext_vector_type(8)));
typedef float    v8f  __attribute__((ext_vector_type(8)));
typedef float    v4f  __attribute__((ext_vector_type(4)));

union Frag16 { v16h v; v8h h[2]; };

// ---------------------------------------------------------------------------
// WMMA GEMM:  C[M x N] = A[M x K] * W[N x K]^T     (f32 in, f16 WMMA, f32 acc)
// 128 threads = 4 waves; tile 64(M) x 16(N), K step 32.
// Waves 0-1 stage A (one full 32-half row per thread, compile-time chunk
// validity); waves 2-3 stage B (two v4f loads + ds_store_b128 per thread).
// ---------------------------------------------------------------------------
template<int K, int N>
__global__ __launch_bounds__(128) void k_gemm_wmma(
    const float* __restrict__ A, int lda,
    const float* __restrict__ W,          // N x K row-major
    float* __restrict__ C, int ldc)
{
  __shared__ __align__(16) _Float16 sA[64][32];
  __shared__ __align__(16) _Float16 sB[16][32];
  const int tid  = threadIdx.x;
  const int wave = tid >> 5;
  const int lane = tid & 31;
  const int m0 = blockIdx.x * 64;
  const int n0 = blockIdx.y * 16;
  constexpr int KT = (K + 31) / 32;

  // staging roles
  const float* arow = A + (size_t)(m0 + tid) * lda;          // tid < 64
  const int tB  = tid & 63;                                  // tid >= 64
  const int rB  = tB & 15;
  const int csB = ((tB >> 4) & 3) * 8;
  const int nB  = n0 + rB;
  const int mr  = lane & 15, ko = (lane >> 4) * 8;

  v8f acc = {};
  #pragma unroll
  for (int kt = 0; kt < KT; ++kt) {
    const int k0 = kt * 32;
    if (tid < 64) {                       // ---- A: full row, all compile-time
      #pragma unroll
      for (int c = 0; c < 4; ++c) {
        const int kb = k0 + c * 8;        // compile-time
        v8h h;
        if (kb + 8 <= K) {                // compile-time (K % 8 == 0)
          v4f f0 = *(const v4f*)(arow + kb);
          v4f f1 = *(const v4f*)(arow + kb + 4);
          #pragma unroll
          for (int j = 0; j < 4; ++j) { h[j] = (_Float16)f0[j]; h[4+j] = (_Float16)f1[j]; }
        } else {
          #pragma unroll
          for (int j = 0; j < 8; ++j) h[j] = (_Float16)0.0f;
        }
        *(v8h*)&sA[tid][c * 8] = h;       // ds_store_b128
      }
    } else {                              // ---- B: 8 halves per thread
      bool valid = (nB < N) && (k0 + csB + 8 <= K);
      const float* wp = W + (valid ? ((size_t)nB * K + k0 + csB) : 0);
      v4f f0 = *(const v4f*)wp;
      v4f f1 = *(const v4f*)(wp + 4);
      v8h h;
      #pragma unroll
      for (int j = 0; j < 4; ++j) {
        h[j]     = (_Float16)(valid ? f0[j] : 0.0f);
        h[4 + j] = (_Float16)(valid ? f1[j] : 0.0f);
      }
      *(v8h*)&sB[rB][csB] = h;
    }
    __syncthreads();
    Frag16 fa, fb;                        // ISA 16-bit A 16x32 layout
    fa.h[0] = *(const v8h*)&sA[wave * 16 + mr][ko];
    fa.h[1] = *(const v8h*)&sA[wave * 16 + mr][ko + 16];
    fb.h[0] = *(const v8h*)&sB[mr][ko];
    fb.h[1] = *(const v8h*)&sB[mr][ko + 16];
    acc = __builtin_amdgcn_wmma_f32_16x16x32_f16(false, fa.v, false, fb.v,
                                                 (short)0, acc, false, false);
    __syncthreads();
  }
  // C/D layout: VGPR r -> m = (lane>>4)*8 + r, n = lane&15
  int n = n0 + (lane & 15);
  if (n < N) {
    float* op = C + (size_t)(m0 + wave * 16 + (lane >> 4) * 8) * ldc + n;
    #pragma unroll
    for (int r = 0; r < 8; ++r) op[r * ldc] = acc[r];
  }
}

// ---------------------------------------------------------------------------
// WMMA implicit-GEMM 3x3 SAME conv.  M = B*HW rows, N = Co = 48, K = Cin*9.
// Waves 0-1 stage A rows: kk -> (ci,dy,dx) fully constant-folded; 9 boundary
// masks + clamped pixel offsets precomputed once per thread.
// ---------------------------------------------------------------------------
template<int Cin, bool InNLC, bool OutNLC, bool HasRes>
__global__ __launch_bounds__(128) void k_conv3x3_wmma(
    const float* __restrict__ In, const float* __restrict__ Wt,
    const float* __restrict__ bias, const float* __restrict__ res,
    float* __restrict__ Out)
{
  constexpr int K  = Cin * 9;
  constexpr int Co = G0C;
  constexpr int KT = (K + 31) / 32;
  __shared__ __align__(16) _Float16 sA[64][32];
  __shared__ __align__(16) _Float16 sB[16][32];
  const int tid  = threadIdx.x;
  const int wave = tid >> 5;
  const int lane = tid & 31;
  const int m0 = blockIdx.x * 64;
  const int n0 = blockIdx.y * 16;
  const int bb = m0 / HWSZ;               // block-uniform (tile never straddles batch)
  const int p0 = m0 - bb * HWSZ;

  // A-staging per-thread precompute (tid < 64): 9 masks + clamped pixel offsets
  int  qcl[9];
  bool msk[9];
  {
    const int p  = p0 + (tid & 63);
    const int hy = p / WWC, wx = p - hy * WWC;
    #pragma unroll
    for (int t9 = 0; t9 < 9; ++t9) {
      const int dy = t9 / 3 - 1, dx = t9 % 3 - 1;   // compile-time
      int yy = hy + dy, xx = wx + dx;
      msk[t9] = ((unsigned)yy < HH) && ((unsigned)xx < WWC);
      int ycl = yy < 0 ? 0 : (yy > HH - 1 ? HH - 1 : yy);
      int xcl = xx < 0 ? 0 : (xx > WWC - 1 ? WWC - 1 : xx);
      qcl[t9] = ycl * WWC + xcl;
    }
  }
  const float* abase = InNLC ? (In + (size_t)bb * HWSZ * Cin)
                             : (In + (size_t)bb * Cin * HWSZ);
  // B-staging role (tid >= 64)
  const int tB  = tid & 63;
  const int rB  = tB & 15;
  const int csB = ((tB >> 4) & 3) * 8;
  const int nB  = n0 + rB;
  const int mr  = lane & 15, ko = (lane >> 4) * 8;

  v8f acc = {};
  #pragma unroll
  for (int kt = 0; kt < KT; ++kt) {
    const int k0 = kt * 32;
    if (tid < 64) {                       // ---- A: im2col row, const offsets
      #pragma unroll
      for (int c = 0; c < 4; ++c) {
        v8h h;
        #pragma unroll
        for (int j = 0; j < 8; ++j) {
          const int kk = k0 + c * 8 + j;  // compile-time
          if (kk < K) {
            const int ci = kk / 9, t9 = kk % 9;     // compile-time
            size_t gi = InNLC ? ((size_t)qcl[t9] * Cin + ci)
                              : ((size_t)ci * HWSZ + qcl[t9]);
            float v = abase[gi];
            h[j] = (_Float16)(msk[t9] ? v : 0.0f);
          } else {
            h[j] = (_Float16)0.0f;
          }
        }
        *(v8h*)&sA[tid][c * 8] = h;       // ds_store_b128
      }
    } else {                              // ---- B: weights (48 x K row-major)
      v8h h;
      #pragma unroll
      for (int j = 0; j < 8; ++j) {
        int kk = k0 + csB + j;
        bool kv = (kk < K);
        float f = Wt[kv ? ((size_t)nB * K + kk) : 0];
        h[j] = (_Float16)(kv ? f : 0.0f);
      }
      *(v8h*)&sB[rB][csB] = h;
    }
    __syncthreads();
    Frag16 fa, fb;
    fa.h[0] = *(const v8h*)&sA[wave * 16 + mr][ko];
    fa.h[1] = *(const v8h*)&sA[wave * 16 + mr][ko + 16];
    fb.h[0] = *(const v8h*)&sB[mr][ko];
    fb.h[1] = *(const v8h*)&sB[mr][ko + 16];
    acc = __builtin_amdgcn_wmma_f32_16x16x32_f16(false, fa.v, false, fb.v,
                                                 (short)0, acc, false, false);
    __syncthreads();
  }
  // store: base pointer per lane, immediate offsets
  const int n  = n0 + (lane & 15);
  const int pb = p0 + wave * 16 + (lane >> 4) * 8;
  const float bn = bias[n];
  if (OutNLC) {
    float* op = Out + ((size_t)bb * HWSZ + pb) * Co + n;
    #pragma unroll
    for (int r = 0; r < 8; ++r) op[r * Co] = acc[r] + bn;
  } else {
    float* op = Out + ((size_t)bb * Co + n) * HWSZ + pb;
    if (HasRes) {
      const float* rp = res + ((size_t)bb * Co + n) * HWSZ + pb;
      #pragma unroll
      for (int r = 0; r < 8; ++r) op[r] = acc[r] + bn + rp[r];
    } else {
      #pragma unroll
      for (int r = 0; r < 8; ++r) op[r] = acc[r] + bn;
    }
  }
}

// ---------------------------------------------------------------------------
// Small VALU kernels
// ---------------------------------------------------------------------------
__global__ void k_add(const float* __restrict__ a, const float* __restrict__ b,
                      float* __restrict__ out, int n)
{
  int i = blockIdx.x * blockDim.x + threadIdx.x;
  if (i >= n) return;
  out[i] = a[i] + (b ? b[i] : 0.0f);
}

__global__ void k_layernorm(const float* __restrict__ in, const float* __restrict__ w,
                            const float* __restrict__ b, float* __restrict__ out, int rows)
{
  int r = blockIdx.x * blockDim.x + threadIdx.x;
  if (r >= rows) return;
  const float* x = in + (size_t)r * G0C;
  float s = 0.f, ss = 0.f;
  #pragma unroll
  for (int c = 0; c < G0C; ++c) { float v = x[c]; s += v; ss += v * v; }
  float mu  = s * (1.0f / G0C);
  float var = ss * (1.0f / G0C) - mu * mu;
  float inv = rsqrtf(var + 1e-5f);
  float* o = out + (size_t)r * G0C;
  #pragma unroll
  for (int c = 0; c < G0C; ++c) o[c] = (x[c] - mu) * inv * w[c] + b[c];
}

// 1x1 conv from NCHW input to (B, HW, Co) output
__global__ void k_conv1x1(const float* __restrict__ in, const float* __restrict__ Wt,
                          const float* __restrict__ bias, float* __restrict__ out,
                          int Cin, int Co)
{
  int idx = blockIdx.x * blockDim.x + threadIdx.x;
  if (idx >= BATCHN * HWSZ * Co) return;
  int co = idx % Co; int t = idx / Co; int p = t % HWSZ; int b = t / HWSZ;
  float acc = bias[co];
  for (int ci = 0; ci < Cin; ++ci)
    acc += Wt[co * Cin + ci] * in[((size_t)b * Cin + ci) * HWSZ + p];
  out[((size_t)b * HWSZ + p) * Co + co] = acc;
}

// causal 4-tap depthwise conv along L + SiLU. in row stride ld (192 or 96).
__global__ void k_dwconv1d_silu(const float* __restrict__ in, int ld,
                                const float* __restrict__ w, const float* __restrict__ bias,
                                float* __restrict__ out)
{
  int idx = blockIdx.x * blockDim.x + threadIdx.x;
  if (idx >= BATCHN * LSEQ * DIN) return;
  int d = idx % DIN; int t = idx / DIN; int l = t % LSEQ; int b = t / LSEQ;
  float acc = bias[d];
  #pragma unroll
  for (int j = 0; j < 4; ++j) {
    int li = l - 3 + j;
    bool inb = (li >= 0);
    float v = in[((size_t)b * LSEQ + (inb ? li : 0)) * ld + d];
    acc += w[d * 4 + j] * (inb ? v : 0.0f);
  }
  out[idx] = acc / (1.0f + __expf(-acc));   // SiLU
}

// dt = softplus(dbl[:, :3] @ dt_w.T + dt_b)
__global__ void k_dt_softplus(const float* __restrict__ dbl, const float* __restrict__ dtw,
                              const float* __restrict__ dtb, float* __restrict__ dt)
{
  int idx = blockIdx.x * blockDim.x + threadIdx.x;
  if (idx >= BATCHN * LSEQ * DIN) return;
  int d = idx % DIN; int rl = idx / DIN;
  const float* row = dbl + (size_t)rl * DBLW;
  float acc = dtb[d] + row[0] * dtw[d * 3 + 0] + row[1] * dtw[d * 3 + 1]
                     + row[2] * dtw[d * 3 + 2];
  dt[idx] = (acc > 20.0f) ? acc : log1pf(__expf(acc));
}

// ---------------------------------------------------------------------------
// Selective scan: 16 lanes per (b,d) own the 16 SSM states in registers.
// backward=1 folds the sequence reversal into the index map.
// ---------------------------------------------------------------------------
__global__ __launch_bounds__(256) void k_scan(
    const float* __restrict__ x,    // (B,L,DIN)
    const float* __restrict__ dt,   // (B,L,DIN)
    const float* __restrict__ Bm,   // dbl+3   (row stride DBLW)
    const float* __restrict__ Cm,   // dbl+19  (row stride DBLW)
    const float* __restrict__ A_log,// (DIN,16)
    const float* __restrict__ Dp,   // (DIN)
    float* __restrict__ y,          // (B,L,DIN)
    int backward)
{
  int gt = blockIdx.x * blockDim.x + threadIdx.x;
  int n  = gt & 15;
  int g  = gt >> 4;
  if (g >= BATCHN * DIN) return;
  int b = g / DIN, d = g - b * DIN;
  const float Adn = -__expf(A_log[d * DSTATE + n]);
  const float Dd  = Dp[d];
  float h = 0.0f;
  const size_t xbase = (size_t)b * LSEQ * DIN + d;
  const size_t pbase = (size_t)b * LSEQ * DBLW;
  for (int s = 0; s < LSEQ; ++s) {
    int i = backward ? (LSEQ - 1 - s) : s;
    if (s + 1 < LSEQ) {
      int in2 = backward ? (LSEQ - 2 - s) : (s + 1);
      __builtin_prefetch(&Bm[pbase + (size_t)in2 * DBLW], 0, 0); // global_prefetch_b8
    }
    float xv  = x [xbase + (size_t)i * DIN];
    float dtv = dt[xbase + (size_t)i * DIN];
    float Bv  = Bm[pbase + (size_t)i * DBLW + n];
    float Cv  = Cm[pbase + (size_t)i * DBLW + n];
    h = __expf(dtv * Adn) * h + (dtv * xv) * Bv;
    float yv = h * Cv;
    yv += __shfl_xor(yv, 1, 32);
    yv += __shfl_xor(yv, 2, 32);
    yv += __shfl_xor(yv, 4, 32);
    yv += __shfl_xor(yv, 8, 32);
    if (n == 0) y[xbase + (size_t)i * DIN] = yv + xv * Dd;
  }
}

// y = (yf + yr) * silu(z),  z = xz[..., 96:192]
__global__ void k_gate(const float* __restrict__ yf, const float* __restrict__ yr,
                       const float* __restrict__ xz, float* __restrict__ y)
{
  int idx = blockIdx.x * blockDim.x + threadIdx.x;
  if (idx >= BATCHN * LSEQ * DIN) return;
  int d = idx % DIN; int rl = idx / DIN;
  float z = xz[(size_t)rl * 192 + 96 + d];
  y[idx] = (yf[idx] + yr[idx]) * (z / (1.0f + __expf(-z)));
}

// depthwise 3x3 SAME conv on NLC tensor g + residual, output NLC
__global__ void k_dwconv3x3_res(const float* __restrict__ g, const float* __restrict__ w,
                                const float* __restrict__ bias, float* __restrict__ out)
{
  int idx = blockIdx.x * blockDim.x + threadIdx.x;
  if (idx >= BATCHN * HWSZ * G0C) return;
  int c = idx % G0C; int t = idx / G0C; int p = t % HWSZ; int b = t / HWSZ;
  int hy = p / WWC, wx = p - hy * WWC;
  float acc = bias[c];
  #pragma unroll
  for (int dy = -1; dy <= 1; ++dy)
    #pragma unroll
    for (int dx = -1; dx <= 1; ++dx) {
      int yy = hy + dy, xx = wx + dx;
      bool inb = ((unsigned)yy < HH) && ((unsigned)xx < WWC);
      size_t gi = ((size_t)b * HWSZ + (inb ? (yy * WWC + xx) : 0)) * G0C + c;
      float v = g[gi];
      acc += w[c * 9 + (dy + 1) * 3 + (dx + 1)] * (inb ? v : 0.0f);
    }
  out[idx] = acc + g[idx];
}

// ---------------------------------------------------------------------------
// Orchestration
// ---------------------------------------------------------------------------
extern "C" void kernel_launch(void* const* d_in, const int* in_sizes, int n_in,
                              void* d_out, int out_size, void* d_ws, size_t ws_size,
                              hipStream_t stream)
{
  (void)in_sizes; (void)n_in; (void)out_size; (void)ws_size;
  auto F = [&](int i) { return (const float*)d_in[i]; };
  const float* pred = F(0);
  const float* ref  = F(1);
  // params flat order: 2..13 top convs, 14..105 blocks (4 x 23), 106..113 tail convs

  float* ws = (float*)d_ws;
  size_t off = 0;
  auto AL = [&](size_t nelem) { float* p = ws + off; off += nelem; return p; };
  const size_t NC = (size_t)BATCHN * G0C * HWSZ;   // 884736

  float* TMP    = AL(NC);
  float* PB     = AL(NC);   // pred_basic (NCHW)
  float* RB     = AL(NC);   // ref_basic
  float* PREDF  = AL(NC);   // (B,L,48)
  float* REFF   = AL(NC);
  float* MSRESI = AL(NC);
  float* T1     = AL(NC);
  float* T2     = AL(NC);   // ms_nn
  float* T3     = AL(NC);   // pan_nn
  float* GB     = AL(NC);   // out_proj result
  float* PF     = AL(NC);   // block output (ms for next block)
  float* XZ     = AL((size_t)MROWS * 192);
  float* XZB    = AL((size_t)MROWS * DIN);
  float* X      = AL((size_t)MROWS * DIN);
  float* XB     = AL((size_t)MROWS * DIN);
  float* DBLF   = AL((size_t)MROWS * DBLW);
  float* DBLB   = AL((size_t)MROWS * DBLW);
  float* DTF    = AL((size_t)MROWS * DIN);
  float* DTB    = AL((size_t)MROWS * DIN);
  float* YF     = AL((size_t)MROWS * DIN);
  float* YR     = AL((size_t)MROWS * DIN);
  float* Y      = AL((size_t)MROWS * DIN);

  const int ETH = 256;
  const int nEl = MROWS * DIN;
  const int nG  = MROWS * G0C;
  dim3 cgrid(MROWS / 64, 3);                // Co=48 -> 3 N-tiles

  // --- shallow feature extraction ---
  k_conv3x3_wmma<3,  false, false, false><<<cgrid, 128, 0, stream>>>(pred, F(2), F(3), nullptr, TMP);
  k_conv3x3_wmma<48, false, false, false><<<cgrid, 128, 0, stream>>>(TMP,  F(4), F(5), nullptr, PB);
  k_conv3x3_wmma<3,  false, false, false><<<cgrid, 128, 0, stream>>>(ref,  F(6), F(7), nullptr, TMP);
  k_conv3x3_wmma<48, false, false, false><<<cgrid, 128, 0, stream>>>(TMP,  F(8), F(9), nullptr, RB);
  k_conv1x1<<<(nG + ETH - 1) / ETH, ETH, 0, stream>>>(PB, F(10), F(11), PREDF, G0C, G0C);
  k_conv1x1<<<(nG + ETH - 1) / ETH, ETH, 0, stream>>>(RB, F(12), F(13), REFF,  G0C, G0C);

  auto cross_mamba = [&](int bi, const float* ms, const float* pan, int first, float* outMs) {
    const int p0 = 14 + bi * 23;
    auto BP = [&](int j) { return F(p0 + j); };
    // ms_resi = ms + resi  (resi == 0 for first block)
    k_add<<<((int)NC + ETH - 1) / ETH, ETH, 0, stream>>>(ms, first ? nullptr : MSRESI,
                                                         MSRESI, (int)NC);
    // double layernorms (as written in the reference)
    k_layernorm<<<(MROWS + ETH - 1) / ETH, ETH, 0, stream>>>(MSRESI, BP(0), BP(1), T1, MROWS);
    k_layernorm<<<(MROWS + ETH - 1) / ETH, ETH, 0, stream>>>(T1,     BP(0), BP(1), T2, MROWS);
    k_layernorm<<<(MROWS + ETH - 1) / ETH, ETH, 0, stream>>>(pan,    BP(2), BP(3), T1, MROWS);
    k_layernorm<<<(MROWS + ETH - 1) / ETH, ETH, 0, stream>>>(T1,     BP(2), BP(3), T3, MROWS);
    // projections (WMMA)
    dim3 g192(MROWS / 64, 12), g96(MROWS / 64, 6), g35(MROWS / 64, 3), g48(MROWS / 64, 3);
    k_gemm_wmma<48, 192><<<g192, 128, 0, stream>>>(T2, G0C, BP(6), XZ,  192); // in_proj
    k_gemm_wmma<48,  96><<<g96,  128, 0, stream>>>(T3, G0C, BP(7), XZB, DIN); // x half only
    // causal dwconv1d + silu
    k_dwconv1d_silu<<<(nEl + ETH - 1) / ETH, ETH, 0, stream>>>(XZ,  192, BP(8),  BP(9),  X);
    k_dwconv1d_silu<<<(nEl + ETH - 1) / ETH, ETH, 0, stream>>>(XZB, DIN, BP(10), BP(11), XB);
    // ssm params (fwd + bwd share x_b; reversal folded into scan index map)
    k_gemm_wmma<96, DBLW><<<g35, 128, 0, stream>>>(XB, DIN, BP(12), DBLF, DBLW);
    k_gemm_wmma<96, DBLW><<<g35, 128, 0, stream>>>(XB, DIN, BP(13), DBLB, DBLW);
    k_dt_softplus<<<(nEl + ETH - 1) / ETH, ETH, 0, stream>>>(DBLF, BP(14), BP(15), DTF);
    k_dt_softplus<<<(nEl + ETH - 1) / ETH, ETH, 0, stream>>>(DBLB, BP(16), BP(17), DTB);
    // selective scans (forward and backward)
    k_scan<<<(BATCHN * DIN * 16) / 256, 256, 0, stream>>>(X, DTF, DBLF + 3, DBLF + 19,
                                                          BP(18), BP(20), YF, 0);
    k_scan<<<(BATCHN * DIN * 16) / 256, 256, 0, stream>>>(X, DTB, DBLB + 3, DBLB + 19,
                                                          BP(19), BP(21), YR, 1);
    // gate + out_proj + depthwise 3x3 residual
    k_gate<<<(nEl + ETH - 1) / ETH, ETH, 0, stream>>>(YF, YR, XZ, Y);
    k_gemm_wmma<96, 48><<<g48, 128, 0, stream>>>(Y, DIN, BP(22), GB, G0C);
    k_dwconv3x3_res<<<(nG + ETH - 1) / ETH, ETH, 0, stream>>>(GB, BP(4), BP(5), outMs);
  };

  // blocks 0 and 1 (pred branch). Blocks 2/3 produce rf, which is dead code
  // in the reference's forward() — legitimately eliminated.
  cross_mamba(0, PREDF, REFF, 1, PF);
  cross_mamba(1, PF,    REFF, 0, PF);

  // --- reconstruction heads ---
  float* OUT0 = (float*)d_out;       // pred_out (NCHW)
  float* OUT1 = OUT0 + NC;           // ref_out
  k_conv3x3_wmma<48, true,  false, false><<<cgrid, 128, 0, stream>>>(PF,   F(106), F(107), nullptr, TMP);
  k_conv3x3_wmma<48, false, false, true ><<<cgrid, 128, 0, stream>>>(TMP,  F(108), F(109), PB, OUT0);
  k_conv3x3_wmma<48, false, false, false><<<cgrid, 128, 0, stream>>>(OUT0, F(110), F(111), nullptr, TMP);
  k_conv3x3_wmma<48, false, false, true ><<<cgrid, 128, 0, stream>>>(TMP,  F(112), F(113), RB, OUT1);
}
